// histoGCN_31937376813166
// MI455X (gfx1250) — matrix operand order repaired and checked by reading, not compiled, not run on hardware
//
#include <hip/hip_runtime.h>
#include <hip/hip_bf16.h>
#include <stdint.h>

typedef __attribute__((ext_vector_type(16))) __bf16 v16bf;
typedef __attribute__((ext_vector_type(8)))  float  v8f;

// ---------------------------------------------------------------- utilities
__global__ void k_fill(float* __restrict__ p, float v, long n) {
  long i = (long)blockIdx.x * blockDim.x + threadIdx.x;
  if (i < n) p[i] = v;
}

__global__ void k_cvt_bf16(const float* __restrict__ in, __bf16* __restrict__ out, long n) {
  long i = (long)blockIdx.x * blockDim.x + threadIdx.x;
  if (i < n) out[i] = (__bf16)in[i];
}

__global__ void k_relu_cvt(const float* __restrict__ in, __bf16* __restrict__ out, long n) {
  long i = (long)blockIdx.x * blockDim.x + threadIdx.x;
  if (i < n) {
    float v = in[i];
    out[i] = (__bf16)(v > 0.0f ? v : 0.0f);
  }
}

// Wt[j*K + k] = (bf16) W[k*Ho + j]  (column-major bf16 copy of the weight)
__global__ void k_cvt_w_t(const float* __restrict__ W, __bf16* __restrict__ Wt, int K, int Ho) {
  int i = blockIdx.x * blockDim.x + threadIdx.x;
  if (i >= K * Ho) return;
  int k = i / Ho, j = i % Ho;
  Wt[(long)j * K + k] = (__bf16)W[i];
}

// ---------------------------------------------------------------- degrees
__global__ void k_deg_edges(const int* __restrict__ dst, float* __restrict__ deg, int E) {
  int e = blockIdx.x * blockDim.x + threadIdx.x;
  if (e < E) atomicAdd(&deg[dst[e]], 1.0f);
}

__global__ void k_rsqrt_ip(float* __restrict__ p, int n) {
  int i = blockIdx.x * blockDim.x + threadIdx.x;
  if (i < n) {
    float d = p[i];
    p[i] = (d > 0.0f) ? rsqrtf(d) : 0.0f;
  }
}

// ---------------------------------------------------------------- WMMA GEMM
// C[M,Ho] = A[M,K] @ B[K,Ho], A row-major bf16, B transposed (Bt[Ho,K] bf16).
// Each wave computes a 16x64 output block (4 WMMA tiles, 4 f32 accumulators),
// so the A fragment is loaded once and reused for 4 v_wmma issues.
// K multiple of 32; M multiple of 16; Ho multiple of 64.
__global__ void k_gemm_wmma(const __bf16* __restrict__ A, const __bf16* __restrict__ Bt,
                            float* __restrict__ C, int K, int Ho, int numTiles) {
  const int wave = threadIdx.x >> 5;
  const int lane = threadIdx.x & 31;
  const int tile = blockIdx.x * (blockDim.x >> 5) + wave;
  if (tile >= numTiles) return;                 // wave-uniform: EXEC stays all-ones
  const int c4tiles = Ho >> 6;                  // 64-wide column blocks
  const int rt = tile / c4tiles;
  const int c4 = tile - rt * c4tiles;
  const int m15  = lane & 15;
  const int half = lane >> 4;

  const __bf16* arow  = A  + (long)(rt * 16 + m15) * K;       // A: row = lane&15
  const __bf16* bbase = Bt + (long)(c4 * 64 + m15) * K;       // B tile j at + j*16*K
  const long bstep = (long)16 * K;                            // elems between col tiles

  v8f acc0 = {}, acc1 = {}, acc2 = {}, acc3 = {};
  for (int k0 = 0; k0 < K; k0 += 32) {
    union { uint4 q[2]; v16bf v; } a, b0, b1, b2, b3;
    // A frag: K = {half*16 .. +15} in elems 0..15
    a.q[0] = *reinterpret_cast<const uint4*>(arow + k0 + half * 16);
    a.q[1] = *reinterpret_cast<const uint4*>(arow + k0 + half * 16 + 8);
    // B frags: K = {half*16 .. +15}, contiguous 32B per tile, constant tile offsets
    const __bf16* bp = bbase + k0 + half * 16;
    b0.q[0] = reinterpret_cast<const uint4*>(bp)[0];
    b0.q[1] = reinterpret_cast<const uint4*>(bp)[1];
    b1.q[0] = reinterpret_cast<const uint4*>(bp + bstep)[0];
    b1.q[1] = reinterpret_cast<const uint4*>(bp + bstep)[1];
    b2.q[0] = reinterpret_cast<const uint4*>(bp + 2 * bstep)[0];
    b2.q[1] = reinterpret_cast<const uint4*>(bp + 2 * bstep)[1];
    b3.q[0] = reinterpret_cast<const uint4*>(bp + 3 * bstep)[0];
    b3.q[1] = reinterpret_cast<const uint4*>(bp + 3 * bstep)[1];
    acc0 = __builtin_amdgcn_wmma_f32_16x16x32_bf16(false, a.v, false, b0.v,
                                                   (short)0, acc0, false, false);
    acc1 = __builtin_amdgcn_wmma_f32_16x16x32_bf16(false, a.v, false, b1.v,
                                                   (short)0, acc1, false, false);
    acc2 = __builtin_amdgcn_wmma_f32_16x16x32_bf16(false, a.v, false, b2.v,
                                                   (short)0, acc2, false, false);
    acc3 = __builtin_amdgcn_wmma_f32_16x16x32_bf16(false, a.v, false, b3.v,
                                                   (short)0, acc3, false, false);
  }
  // D: VGPR r -> row half*8 + r, col lane&15 (+ 16 per tile)
  float* crow = C + (long)(rt * 16 + half * 8) * Ho + c4 * 64 + m15;
#pragma unroll
  for (int r = 0; r < 8; ++r) {
    crow[(long)r * Ho +  0] = acc0[r];
    crow[(long)r * Ho + 16] = acc1[r];
    crow[(long)r * Ho + 32] = acc2[r];
    crow[(long)r * Ho + 48] = acc3[r];
  }
}

// ---------------------------------------------------------------- aggregation (H = 256)
// agg[i,:] = bias + dinv[i]^2 * m[i,:]   (self-loop term folded in)
__global__ void k_agg_init(const float* __restrict__ m, const float* __restrict__ dinv,
                           const float* __restrict__ bias, float* __restrict__ agg) {
  int i = blockIdx.x, c = threadIdx.x;          // blockDim = 256
  float di = dinv[i];
  agg[(long)i * 256 + c] = bias[c] + di * di * m[(long)i * 256 + c];
}

// agg[dst,:] += dinv[src]*dinv[dst] * m[src,:]; 64 lanes x float4 per edge, 4 edges/block
__global__ void k_agg_edges(const int* __restrict__ src, const int* __restrict__ dst,
                            const float* __restrict__ dinv, const float* __restrict__ m,
                            float* __restrict__ agg, int E) {
  int e = blockIdx.x * 4 + (threadIdx.x >> 6);
  if (e >= E) return;
  int s = src[e], d = dst[e];
  float nrm = dinv[s] * dinv[d];
  int c = (threadIdx.x & 63) * 4;
  float4 v = *reinterpret_cast<const float4*>(m + (long)s * 256 + c);
  float* ap = agg + (long)d * 256 + c;
  atomicAdd(ap + 0, nrm * v.x);
  atomicAdd(ap + 1, nrm * v.y);
  atomicAdd(ap + 2, nrm * v.z);
  atomicAdd(ap + 3, nrm * v.w);
}

// ---------------------------------------------------------------- pooling + head
__global__ void k_count(const int* __restrict__ batch, float* __restrict__ counts, int N) {
  int i = blockIdx.x * blockDim.x + threadIdx.x;
  if (i < N) atomicAdd(&counts[batch[i]], 1.0f);
}

__global__ void k_pool_sum(const float* __restrict__ h, const int* __restrict__ batch,
                           float* __restrict__ pooled) {
  int i = blockIdx.x, c = threadIdx.x;          // blockDim = 256
  atomicAdd(&pooled[(long)batch[i] * 256 + c], h[(long)i * 256 + c]);
}

__global__ void k_pool_div(float* __restrict__ pooled, const float* __restrict__ counts) {
  int g = blockIdx.x, c = threadIdx.x;          // grid = G, blockDim = 256
  pooled[g * 256 + c] /= fmaxf(counts[g], 1.0f);
}

__global__ void k_head1(const float* __restrict__ pooled, const float* __restrict__ Wf,
                        const float* __restrict__ bfv, float* __restrict__ fc1) {
  int t = blockIdx.x * blockDim.x + threadIdx.x;   // 16*64 outputs
  if (t >= 16 * 64) return;
  int g = t >> 6, j = t & 63;
  float s = bfv[j];
  for (int c = 0; c < 256; ++c) s += pooled[g * 256 + c] * Wf[c * 64 + j];
  fc1[t] = s;
}

__global__ void k_head2(const float* __restrict__ fc1, const float* __restrict__ Wp,
                        const float* __restrict__ bp, float* __restrict__ out) {
  int g = threadIdx.x;
  if (g >= 16) return;
  float s = bp[0];
  for (int j = 0; j < 64; ++j) s += fc1[g * 64 + j] * Wp[j];
  out[g] = 1.0f / (1.0f + expf(-s));
}

// ---------------------------------------------------------------- launcher
extern "C" void kernel_launch(void* const* d_in, const int* in_sizes, int n_in,
                              void* d_out, int out_size, void* d_ws, size_t ws_size,
                              hipStream_t stream) {
  const int D = 512, H = 256, G = 16;
  const int N = in_sizes[0] / D;
  const int E = in_sizes[1] / 2;

  const float* x   = (const float*)d_in[0];
  const int*   ei  = (const int*)d_in[1];
  const int*   bat = (const int*)d_in[2];
  const float* W1  = (const float*)d_in[3];  const float* b1 = (const float*)d_in[4];
  const float* W2  = (const float*)d_in[5];  const float* b2 = (const float*)d_in[6];
  const float* W3  = (const float*)d_in[7];  const float* b3 = (const float*)d_in[8];
  const float* Wf  = (const float*)d_in[9];  const float* bf = (const float*)d_in[10];
  const float* Wp  = (const float*)d_in[11]; const float* bp = (const float*)d_in[12];
  const int* src = ei;
  const int* dst = ei + E;
  (void)n_in; (void)out_size; (void)ws_size;

  // workspace carve-out
  char* ws = (char*)d_ws;
  size_t off = 0;
  auto take = [&](size_t bytes) -> char* {
    char* p = ws + off;
    off += (bytes + 255) & ~(size_t)255;
    return p;
  };
  __bf16* hbf    = (__bf16*)take((size_t)N * D * 2);   // bf16 activations (reused all layers)
  float*  mbuf   = (float*) take((size_t)N * H * 4);   // GEMM output m = h @ W
  float*  agg    = (float*) take((size_t)N * H * 4);   // aggregated output
  __bf16* wt     = (__bf16*)take((size_t)D * H * 2);   // transposed bf16 weight (reused)
  float*  dinv   = (float*) take((size_t)N * 4);
  float*  pooled = (float*) take((size_t)G * H * 4);
  float*  counts = (float*) take((size_t)G * 4);
  float*  fc1    = (float*) take((size_t)G * 64 * 4);

  const int tiles = (N / 16) * (H / 64);               // 16x64 wave-tiles
  const int gemmBlocks = (tiles + 7) / 8;              // 8 waves per 256-thread block
  const long nd = (long)N * D, nh = (long)N * H;

  // --- degrees: deg = 1 (self loop) + in-degree; then dinv = rsqrt(deg)
  k_fill<<<(N + 255) / 256, 256, 0, stream>>>(dinv, 1.0f, N);
  k_deg_edges<<<(E + 255) / 256, 256, 0, stream>>>(dst, dinv, E);
  k_rsqrt_ip<<<(N + 255) / 256, 256, 0, stream>>>(dinv, N);

  // --- layer 1: m = x @ W1 ; agg = scatter + bias ; h = relu(agg) -> bf16
  k_cvt_bf16<<<(int)((nd + 255) / 256), 256, 0, stream>>>(x, hbf, nd);
  k_cvt_w_t<<<(D * H + 255) / 256, 256, 0, stream>>>(W1, wt, D, H);
  k_gemm_wmma<<<gemmBlocks, 256, 0, stream>>>(hbf, wt, mbuf, D, H, tiles);
  k_agg_init<<<N, 256, 0, stream>>>(mbuf, dinv, b1, agg);
  k_agg_edges<<<(E + 3) / 4, 256, 0, stream>>>(src, dst, dinv, mbuf, agg, E);
  k_relu_cvt<<<(int)((nh + 255) / 256), 256, 0, stream>>>(agg, hbf, nh);

  // --- layer 2
  k_cvt_w_t<<<(H * H + 255) / 256, 256, 0, stream>>>(W2, wt, H, H);
  k_gemm_wmma<<<gemmBlocks, 256, 0, stream>>>(hbf, wt, mbuf, H, H, tiles);
  k_agg_init<<<N, 256, 0, stream>>>(mbuf, dinv, b2, agg);
  k_agg_edges<<<(E + 3) / 4, 256, 0, stream>>>(src, dst, dinv, mbuf, agg, E);
  k_relu_cvt<<<(int)((nh + 255) / 256), 256, 0, stream>>>(agg, hbf, nh);

  // --- layer 3 (no relu)
  k_cvt_w_t<<<(H * H + 255) / 256, 256, 0, stream>>>(W3, wt, H, H);
  k_gemm_wmma<<<gemmBlocks, 256, 0, stream>>>(hbf, wt, mbuf, H, H, tiles);
  k_agg_init<<<N, 256, 0, stream>>>(mbuf, dinv, b3, agg);
  k_agg_edges<<<(E + 3) / 4, 256, 0, stream>>>(src, dst, dinv, mbuf, agg, E);

  // --- mean pool + head
  k_fill<<<(G * H + 255) / 256, 256, 0, stream>>>(pooled, 0.0f, G * H);
  k_fill<<<1, 256, 0, stream>>>(counts, 0.0f, G);
  k_count<<<(N + 255) / 256, 256, 0, stream>>>(bat, counts, N);
  k_pool_sum<<<N, 256, 0, stream>>>(agg, bat, pooled);
  k_pool_div<<<G, 256, 0, stream>>>(pooled, counts);
  k_head1<<<4, 256, 0, stream>>>(pooled, Wf, bf, fc1);
  k_head2<<<1, 32, 0, stream>>>(fc1, Wp, bp, (float*)d_out);
}
